// XdiffMultiHeadAttention_13829794693594
// MI455X (gfx1250) — matrix operand order repaired and checked by reading, not compiled, not run on hardware
//
#include <hip/hip_runtime.h>
#include <hip/hip_bf16.h>
#include <math.h>

// Problem constants (match reference)
#define BB   4
#define SEQ  2048
#define DM   512
#define DA   512
#define NH   8
#define NF   4
#define DEP  64        // DA / NH
#define QTILES (SEQ/16)

typedef __attribute__((ext_vector_type(16))) __bf16 bf16x16;
typedef __attribute__((ext_vector_type(8)))  float  f32x8;
typedef __attribute__((ext_vector_type(4)))  float  f32x4;   // native vec for nontemporal

union FragBF {               // one WMMA bf16 operand (16 bf16 = 32 bytes)
  bf16x16 f;
  uint4   q[2];
};

static __device__ __forceinline__ f32x8 wmma_bf16(const bf16x16& a,
                                                  const bf16x16& b,
                                                  const f32x8& c) {
  return __builtin_amdgcn_wmma_f32_16x16x32_bf16(false, a, false, b,
                                                 (short)0, c, false, false);
}

// ---------------------------------------------------------------------------
// fp32 -> bf16 convert, 4 elements / thread (b128 in, b64 out)
// ---------------------------------------------------------------------------
__global__ void cvt_f32_to_bf16_v4(const float* __restrict__ src,
                                   __bf16* __restrict__ dst, int n4) {
  for (int i = blockIdx.x * blockDim.x + threadIdx.x; i < n4;
       i += gridDim.x * blockDim.x) {
    float4 f = ((const float4*)src)[i];
    union { __bf16 b[4]; uint2 u; } o;
    o.b[0] = (__bf16)f.x; o.b[1] = (__bf16)f.y;
    o.b[2] = (__bf16)f.z; o.b[3] = (__bf16)f.w;
    ((uint2*)dst)[i] = o.u;
  }
}

// ---------------------------------------------------------------------------
// Pack a row-major [K,N] matrix into per-lane WMMA B-fragment layout:
//   Wp[((c*(N/16) + t)*32 + lane)*16 + j] = W[(c*32 + kB(lane) + j + (j&8))*N + t*16 + (lane&15)]
// so a lane's 16 bf16 are contiguous (two b128 loads, coalesced per wave).
// ---------------------------------------------------------------------------
__global__ void pack_bfrag_f32(const float* __restrict__ W,
                               __bf16* __restrict__ Wp, int K, int N) {
  int ntil = N >> 4;
  int total = (K >> 5) * ntil * 32;
  for (int idx = blockIdx.x * blockDim.x + threadIdx.x; idx < total;
       idx += gridDim.x * blockDim.x) {
    int lane = idx & 31;
    int ct   = idx >> 5;            // c*ntil + t
    int ncol = lane & 15;
    int kB   = (lane >> 4) * 8;
    int t    = ct % ntil;
    int c    = ct / ntil;
    const float* src = W + (size_t)(c * 32 + kB) * N + t * 16 + ncol;
    __bf16* dst = Wp + (size_t)idx * 16;
#pragma unroll
    for (int j = 0; j < 16; ++j) dst[j] = (__bf16)src[(size_t)(j + (j & 8)) * N];
  }
}

__global__ void pack_bfrag_bf16(const __bf16* __restrict__ W,
                                __bf16* __restrict__ Wp, int K, int N) {
  int ntil = N >> 4;
  int total = (K >> 5) * ntil * 32;
  for (int idx = blockIdx.x * blockDim.x + threadIdx.x; idx < total;
       idx += gridDim.x * blockDim.x) {
    int lane = idx & 31;
    int ct   = idx >> 5;
    int ncol = lane & 15;
    int kB   = (lane >> 4) * 8;
    int t    = ct % ntil;
    int c    = ct / ntil;
    const __bf16* src = W + (size_t)(c * 32 + kB) * N + t * 16 + ncol;
    __bf16* dst = Wp + (size_t)idx * 16;
#pragma unroll
    for (int j = 0; j < 16; ++j) dst[j] = src[(size_t)(j + (j & 8)) * N];
  }
}

// ---------------------------------------------------------------------------
// WMMA GEMM with pre-packed B: C[M,N] = A[M,K](bf16 rm) * Wp + bias[N].
// One wave computes a 32x32 C tile (2x2 WMMA tiles, 4 WMMAs per k-chunk).
// A frags: two b128 loads per 16x32 frag; B frags: two b128 loads, coalesced.
// ---------------------------------------------------------------------------
template <bool F32OUT>
__global__ void wmma_gemm_packed(const __bf16* __restrict__ A,
                                 const __bf16* __restrict__ Wp,
                                 const float* __restrict__ bias,
                                 void* __restrict__ Cout,
                                 int M, int N, int K) {
  int wave = blockIdx.x * (blockDim.x >> 5) + (threadIdx.x >> 5);
  int Nt = N >> 5;
  int total = (M >> 5) * Nt;
  if (wave >= total) return;
  int mt = wave / Nt;
  int nt = wave - mt * Nt;

  int lane = threadIdx.x & 31;
  int mrow = lane & 15;
  int half = lane >> 4;
  int kA   = half * 8;
  int ncol = lane & 15;
  int ntil = N >> 4;

  const __bf16* Ab0 = A + (size_t)(mt * 32 + mrow) * K;
  const __bf16* Ab1 = Ab0 + (size_t)16 * K;

  f32x8 acc00 = {}, acc01 = {}, acc10 = {}, acc11 = {};
  for (int k0 = 0; k0 < K; k0 += 32) {
    FragBF a0, a1, b0, b1;
    a0.q[0] = *(const uint4*)(Ab0 + k0 + kA);
    a0.q[1] = *(const uint4*)(Ab0 + k0 + kA + 16);
    a1.q[0] = *(const uint4*)(Ab1 + k0 + kA);
    a1.q[1] = *(const uint4*)(Ab1 + k0 + kA + 16);
    const uint4* wp = (const uint4*)(Wp +
        (((size_t)(k0 >> 5) * ntil + nt * 2) * 32 + lane) * 16);
    b0.q[0] = wp[0];  b0.q[1] = wp[1];          // n-tile nt*2
    b1.q[0] = wp[64]; b1.q[1] = wp[65];         // n-tile nt*2+1 (+512 bf16)
    acc00 = wmma_bf16(a0.f, b0.f, acc00);
    acc01 = wmma_bf16(a0.f, b1.f, acc01);
    acc10 = wmma_bf16(a1.f, b0.f, acc10);
    acc11 = wmma_bf16(a1.f, b1.f, acc11);
  }

  float* Cf  = (float*)Cout;
  __bf16* Cb = (__bf16*)Cout;
#pragma unroll
  for (int im = 0; im < 2; ++im) {
#pragma unroll
    for (int in = 0; in < 2; ++in) {
      f32x8 c = (im == 0) ? (in == 0 ? acc00 : acc01)
                          : (in == 0 ? acc10 : acc11);
      int gn = nt * 32 + in * 16 + ncol;
      float bn = bias[gn];
      int gm0 = mt * 32 + im * 16 + (half << 3);
#pragma unroll
      for (int i = 0; i < 8; ++i) {
        size_t idx = (size_t)(gm0 + i) * N + gn;
        float v = c[i] + bn;
        if constexpr (F32OUT) Cf[idx] = v;
        else                  Cb[idx] = (__bf16)v;
      }
    }
  }
}

// ---------------------------------------------------------------------------
// Fused attention: 4 waves per block, one block per (b, h, 16-query tile).
// LDS: 16 x 2048 fp32 logits (128 KB).
//  Pass 1 (waves split key tiles): logits = QK^T/8 + (xdiff . x)/2
//          Q frags + x-vectors hoisted; xdiff read non-temporally (b128).
//  Pass 2 (waves split rows): exact softmax, fully 4-wide vectorized;
//          attn streamed out with b128 non-temporal stores.
//  Pass 3 (wave w owns V column tile w): ctx = attn @ V from packed V frags.
// ---------------------------------------------------------------------------
__global__ void attn_fused(const __bf16* __restrict__ qp,   // [B,S,DA] bf16
                           const __bf16* __restrict__ kp,   // [B,S,DA] bf16
                           const __bf16* __restrict__ vpk,  // packed V frags
                           const float*  __restrict__ xg,   // [B,S,H*F] fp32
                           const float*  __restrict__ xdiff,// [B,S,S,F] fp32
                           float*  __restrict__ attn_out,   // [B,H,S,S] fp32
                           __bf16* __restrict__ ctx_out) {  // [B,S,DA] bf16
  extern __shared__ float lg[];                 // [16][SEQ]

  int wg = blockIdx.x;
  int qt = wg & (QTILES - 1);
  int h  = (wg >> 7) & (NH - 1);
  int b  = wg >> 10;
  int q0 = qt * 16;

  int wv   = threadIdx.x >> 5;                  // wave 0..3
  int lane = threadIdx.x & 31;
  int mrow = lane & 15;
  int half = lane >> 4;
  int kA   = half * 8;
  int kB   = half * 8;
  int ncol = lane & 15;

  // Hoisted Q fragments (two k-chunks of 32 over DEP=64)
  const __bf16* Qb = qp + (size_t)(b * SEQ + q0 + mrow) * DA + h * DEP;
  FragBF aq0, aq1;
  aq0.q[0] = *(const uint4*)(Qb + kA);
  aq0.q[1] = *(const uint4*)(Qb + kA + 16);
  aq1.q[0] = *(const uint4*)(Qb + 32 + kA);
  aq1.q[1] = *(const uint4*)(Qb + 32 + kA + 16);

  // Hoisted x vectors for the 8 C-rows this lane produces
  f32x4 xv[8];
#pragma unroll
  for (int i = 0; i < 8; ++i) {
    int q = q0 + i + (half << 3);
    xv[i] = *(const f32x4*)(xg + (size_t)(b * SEQ + q) * (NH * NF) + h * NF);
  }

  // ---- Pass 1 ----
  for (int kt = wv; kt < QTILES; kt += 4) {
    int k0 = kt * 16;
    const __bf16* Kb = kp + (size_t)(b * SEQ + k0 + ncol) * DA + h * DEP + kB;
    FragBF bk0, bk1;
    bk0.q[0] = *(const uint4*)(Kb);
    bk0.q[1] = *(const uint4*)(Kb + 16);
    bk1.q[0] = *(const uint4*)(Kb + 32);
    bk1.q[1] = *(const uint4*)(Kb + 48);
    f32x8 acc = {};
    acc = wmma_bf16(aq0.f, bk0.f, acc);
    acc = wmma_bf16(aq1.f, bk1.f, acc);
#pragma unroll
    for (int i = 0; i < 8; ++i) {
      int m  = i + (half << 3);
      int kk = k0 + ncol;
      const f32x4* xdp = (const f32x4*)(xdiff +
          (((size_t)b * SEQ + (q0 + m)) * SEQ + kk) * NF);
      f32x4 xd = __builtin_nontemporal_load(xdp);
      float bias = (xd.x * xv[i].x + xd.y * xv[i].y +
                    xd.z * xv[i].z + xd.w * xv[i].w) * 0.5f;  // 1/sqrt(F)
      lg[m * SEQ + kk] = acc[i] * 0.125f + bias;              // 1/sqrt(depth)
    }
  }
  __syncthreads();

  // ---- Pass 2: softmax (4-wide vectorized), each wave owns 4 rows ----
  for (int m = wv * 4; m < wv * 4 + 4; ++m) {
    float* row = lg + m * SEQ;
    float mx = -3.0e38f;
#pragma unroll
    for (int i = 0; i < SEQ / 128; ++i) {           // 16 iters, b128 LDS reads
      f32x4 v4 = *(const f32x4*)(row + i * 128 + lane * 4);
      mx = fmaxf(mx, fmaxf(fmaxf(v4.x, v4.y), fmaxf(v4.z, v4.w)));
    }
#pragma unroll
    for (int o = 16; o > 0; o >>= 1) mx = fmaxf(mx, __shfl_xor(mx, o, 32));
    float sum = 0.0f;
#pragma unroll
    for (int i = 0; i < SEQ / 128; ++i) {
      float* p = row + i * 128 + lane * 4;
      f32x4 v4 = *(const f32x4*)p;
      v4.x = __expf(v4.x - mx); v4.y = __expf(v4.y - mx);
      v4.z = __expf(v4.z - mx); v4.w = __expf(v4.w - mx);
      *(f32x4*)p = v4;
      sum += (v4.x + v4.y) + (v4.z + v4.w);
    }
#pragma unroll
    for (int o = 16; o > 0; o >>= 1) sum += __shfl_xor(sum, o, 32);
    float inv = 1.0f / sum;
    float* arow = attn_out + (((size_t)b * NH + h) * SEQ + (q0 + m)) * SEQ;
#pragma unroll
    for (int i = 0; i < SEQ / 128; ++i) {
      float* p = row + i * 128 + lane * 4;
      f32x4 v4 = *(const f32x4*)p;
      v4.x *= inv; v4.y *= inv; v4.z *= inv; v4.w *= inv;
      *(f32x4*)p = v4;
      __builtin_nontemporal_store(v4, (f32x4*)(arow + i * 128 + lane * 4));
    }
  }
  __syncthreads();

  // ---- Pass 3: ctx = attn @ V ; wave w owns V 16-col tile nt = w ----
  f32x8 o = {};
  int gcbase = b * (SEQ / 32);                  // global 32-row chunk base
  int vt = h * 4 + wv;                          // 16-col tile in DA
  for (int c = 0; c < SEQ / 32; ++c) {
    // A fragment from LDS probabilities (fp32 -> bf16)
    const float* lrow = lg + mrow * SEQ + c * 32;
    f32x4 f0 = *(const f32x4*)(lrow + kA);
    f32x4 f1 = *(const f32x4*)(lrow + kA + 4);
    f32x4 f2 = *(const f32x4*)(lrow + kA + 16);
    f32x4 f3 = *(const f32x4*)(lrow + kA + 20);
    bf16x16 a;
    a[0]  = (__bf16)f0.x; a[1]  = (__bf16)f0.y; a[2]  = (__bf16)f0.z; a[3]  = (__bf16)f0.w;
    a[4]  = (__bf16)f1.x; a[5]  = (__bf16)f1.y; a[6]  = (__bf16)f1.z; a[7]  = (__bf16)f1.w;
    a[8]  = (__bf16)f2.x; a[9]  = (__bf16)f2.y; a[10] = (__bf16)f2.z; a[11] = (__bf16)f2.w;
    a[12] = (__bf16)f3.x; a[13] = (__bf16)f3.y; a[14] = (__bf16)f3.z; a[15] = (__bf16)f3.w;
    // B fragment from packed V
    const uint4* vp4 = (const uint4*)(vpk +
        ((((size_t)(gcbase + c) * (DA / 16)) + vt) * 32 + lane) * 16);
    FragBF bv_;
    bv_.q[0] = vp4[0]; bv_.q[1] = vp4[1];
    o = wmma_bf16(a, bv_.f, o);
  }
#pragma unroll
  for (int i = 0; i < 8; ++i) {
    int m = i + (half << 3);
    ctx_out[(size_t)(b * SEQ + q0 + m) * DA + h * DEP + wv * 16 + ncol] =
        (__bf16)o[i];
  }
}

// ---------------------------------------------------------------------------
// Host launch
// ---------------------------------------------------------------------------
extern "C" void kernel_launch(void* const* d_in, const int* in_sizes, int n_in,
                              void* d_out, int out_size, void* d_ws, size_t ws_size,
                              hipStream_t stream) {
  const float* q     = (const float*)d_in[0];
  const float* k     = (const float*)d_in[1];
  const float* v     = (const float*)d_in[2];
  const float* xdiff = (const float*)d_in[3];
  const float* wq    = (const float*)d_in[4];
  const float* bq    = (const float*)d_in[5];
  const float* wk    = (const float*)d_in[6];
  const float* bk    = (const float*)d_in[7];
  const float* wv_   = (const float*)d_in[8];
  const float* bv    = (const float*)d_in[9];
  const float* wx    = (const float*)d_in[10];
  const float* bx    = (const float*)d_in[11];
  const float* wo    = (const float*)d_in[12];
  const float* bo    = (const float*)d_in[13];

  const size_t BS = (size_t)BB * SEQ;          // 8192 rows
  float* out_part  = (float*)d_out;            // [B,S,DM]
  float* attn_part = (float*)d_out + BS * DM;  // [B,H,S,S]

  char* ws = (char*)d_ws;
  size_t off = 0;
  auto carve = [&](size_t bytes) {
    void* p = ws + off;
    off = (off + bytes + 255) & ~(size_t)255;
    return p;
  };
  __bf16* qbf  = (__bf16*)carve(BS * DM * 2);
  __bf16* kbf  = (__bf16*)carve(BS * DM * 2);
  __bf16* vbf  = (__bf16*)carve(BS * DM * 2);
  __bf16* wqp  = (__bf16*)carve((size_t)DM * DA * 2);   // packed B frags
  __bf16* wkp  = (__bf16*)carve((size_t)DM * DA * 2);
  __bf16* wvp  = (__bf16*)carve((size_t)DM * DA * 2);
  __bf16* wop  = (__bf16*)carve((size_t)DA * DM * 2);
  __bf16* wxp  = (__bf16*)carve((size_t)DA * NH * NF * 2);
  __bf16* qpb  = (__bf16*)carve(BS * DA * 2);
  __bf16* kpb  = (__bf16*)carve(BS * DA * 2);
  __bf16* vpb  = (__bf16*)carve(BS * DA * 2);
  __bf16* vpk  = (__bf16*)carve(BS * DA * 2);           // packed V frags
  float*  xf   = (float*)carve(BS * NH * NF * 4);
  __bf16* ctxb = (__bf16*)carve(BS * DA * 2);

  // 1) activation conversions (vectorized)
  auto cvt = [&](const float* s, __bf16* d, size_t n) {
    int n4 = (int)(n >> 2);
    int blk = 256, grid = (n4 + blk - 1) / blk;
    if (grid > 4096) grid = 4096;
    cvt_f32_to_bf16_v4<<<grid, blk, 0, stream>>>(s, d, n4);
  };
  cvt(q, qbf, BS * DM);
  cvt(k, kbf, BS * DM);
  cvt(v, vbf, BS * DM);

  // 2) pack weights straight from fp32 into B-fragment layout
  auto packw = [&](const float* s, __bf16* d, int K, int N) {
    int total = (K >> 5) * (N >> 4) * 32;
    int blk = 256, grid = (total + blk - 1) / blk;
    pack_bfrag_f32<<<grid, blk, 0, stream>>>(s, d, K, N);
  };
  packw(wq, wqp, DM, DA);
  packw(wk, wkp, DM, DA);
  packw(wv_, wvp, DM, DA);
  packw(wo, wop, DA, DM);
  packw(wx, wxp, DA, NH * NF);

  // 3) projections (bf16 out): 32x32 tile per wave
  {
    int waves = ((int)BS / 32) * (DA / 32);    // 4096
    int blk = 128, grid = (waves + 3) / 4;
    wmma_gemm_packed<false><<<grid, blk, 0, stream>>>(qbf, wqp, bq, qpb,
                                                      (int)BS, DA, DM);
    wmma_gemm_packed<false><<<grid, blk, 0, stream>>>(kbf, wkp, bk, kpb,
                                                      (int)BS, DA, DM);
    wmma_gemm_packed<false><<<grid, blk, 0, stream>>>(vbf, wvp, bv, vpb,
                                                      (int)BS, DA, DM);
  }

  // 4) pack V projection into B-fragment layout for pass 3
  {
    int total = ((int)BS >> 5) * (DA >> 4) * 32;
    int blk = 256, grid = (total + blk - 1) / blk;
    pack_bfrag_bf16<<<grid, blk, 0, stream>>>(vpb, vpk, (int)BS, DA);
  }

  // 5) x = qp @ wx + bx  (fp32 out, N = 32)
  {
    int waves = ((int)BS / 32) * ((NH * NF) / 32);
    int blk = 128, grid = (waves + 3) / 4;
    wmma_gemm_packed<true><<<grid, blk, 0, stream>>>(qpb, wxp, bx, xf,
                                                     (int)BS, NH * NF, DA);
  }

  // 6) fused bias + softmax + attn@V
  {
    int grid = BB * NH * QTILES;                  // 4096 blocks
    size_t lds = (size_t)16 * SEQ * sizeof(float);// 128 KB
    attn_fused<<<grid, 128, lds, stream>>>(qpb, kpb, vpk, xf, xdiff,
                                           attn_part, ctxb);
  }

  // 7) out = ctx @ wo + bo  (fp32 out)
  {
    int waves = ((int)BS / 32) * (DM / 32);
    int blk = 128, grid = (waves + 3) / 4;
    wmma_gemm_packed<true><<<grid, blk, 0, stream>>>(ctxb, wop, bo, out_part,
                                                     (int)BS, DM, DA);
  }
}
